// ThreeBodyLayer_82635170775664
// MI455X (gfx1250) — matrix operand order repaired and checked by reading, not compiled, not run on hardware
//
#include <hip/hip_runtime.h>
#include <hip/hip_bf16.h>

typedef __attribute__((ext_vector_type(16))) _Float16 v16h;
typedef __attribute__((ext_vector_type(8)))  _Float16 v8h;
typedef __attribute__((ext_vector_type(8)))  float    v8f;

// 15 pairs (i<j) x 2 directions. CI feeds the W1b slot, CJ feeds the W1c slot.
__constant__ int CIa[30] = {0,0,0,0,0,1,1,1,1,2,2,2,3,3,4,  1,2,3,4,5,2,3,4,5,3,4,5,4,5,5};
__constant__ int CJa[30] = {1,2,3,4,5,2,3,4,5,3,4,5,4,5,5,  0,0,0,0,0,1,1,1,1,2,2,2,3,3,4};

__device__ __forceinline__ float softplusf(float x) {
  // stable: max(x,0) + log(1 + exp(-|x|))
  float e = __expf(-fabsf(x));
  return fmaxf(x, 0.0f) + __logf(1.0f + e);
}

__device__ __forceinline__ v8f wmma16(v16h a, v16h b, v8f c) {
  return __builtin_amdgcn_wmma_f32_16x16x32_f16(
      /*neg_a=*/false, a, /*neg_b=*/false, b,
      /*c_mod=*/(short)0, c, /*reuse_a=*/false, /*reuse_b=*/false);
}

// ---------------------------------------------------------------------------
// One-shot setup: swizzle W1 (192x64) and W2 (64x32) into WMMA B-fragment
// layout in global scratch. Fragment fid: 1KB = 32 lanes x 16 f16.
//   fid 0..23 : W1 part p=fid>>3 (a/b/c), kt=(fid>>2)&1, nt=fid&3
//   fid 24..27: W2 kt=(fid-24)>>1, nt=(fid-24)&1
// B-frag element e of lane L: column n = nt*16+(L&15), K = kt*32+8*(L>>4)+e+(e&8).
// ---------------------------------------------------------------------------
__global__ __launch_bounds__(32)
void swizzle_weights(const float* __restrict__ W1, const float* __restrict__ W2,
                     _Float16* __restrict__ wg) {
  const int lane = threadIdx.x;
  const int nl = lane & 15, h = lane >> 4;
  const int fid = blockIdx.x;
  _Float16* dst = wg + fid * 512 + lane * 16;
  if (fid < 24) {
    const int p = fid >> 3, kt = (fid >> 2) & 1, nt = fid & 3;
    const int n = nt * 16 + nl;
    #pragma unroll
    for (int e = 0; e < 16; ++e)
      dst[e] = (_Float16)W1[(p * 64 + kt * 32 + 8 * h + e + (e & 8)) * 64 + n];
  } else {
    const int f = fid - 24, kt = f >> 1, nt = f & 1;
    const int n = nt * 16 + nl;
    #pragma unroll
    for (int e = 0; e < 16; ++e)
      dst[e] = (_Float16)W2[(kt * 32 + 8 * h + e + (e & 8)) * 32 + n];
  }
}

// ---------------------------------------------------------------------------
// Main kernel: one wave32 per 16-batch tile.
//   stage 1: u = core@W1a + b1, p_n = lig_n@W1b, q_n = lig_n@W1c  (13 GEMMs)
//   stage 2: 30x [ h1 = softplus(u+p_i+q_j) -> @W2 -> softplus -> dot W3 ]
// USE_WS: B-fragments read straight from pre-swizzled global scratch
// (L2-resident); otherwise swizzled into LDS per block (fallback).
// ---------------------------------------------------------------------------
template <bool USE_WS>
__global__ __launch_bounds__(32)
void three_body_kernel(const float* __restrict__ core, const float* __restrict__ ligs,
                       const float* __restrict__ W1,   const float* __restrict__ b1,
                       const float* __restrict__ W2,   const float* __restrict__ b2,
                       const float* __restrict__ W3,   const float* __restrict__ b3,
                       const _Float16* __restrict__ wg,
                       float* __restrict__ out)
{
  __shared__ _Float16 wfrag[USE_WS ? 16 : 28 * 512];
  // upq: U (16x64) + P[6] + Q[6], f16 row-major [m][k] -- 26KB
  __shared__ _Float16 upq[13 * 1024];

  const int lane = threadIdx.x;   // one wave32 per block
  const int nl   = lane & 15;
  const int h    = lane >> 4;

  auto LDWf = [&](int fid) -> v16h {
    if constexpr (USE_WS)
      return *(const v16h*)(wg + fid * 512 + lane * 16);
    else
      return *(const v16h*)&wfrag[fid * 512 + lane * 16];
  };

  if constexpr (!USE_WS) {
    // in-LDS weight swizzle fallback
    #pragma unroll
    for (int p = 0; p < 3; ++p)
      #pragma unroll
      for (int kt = 0; kt < 2; ++kt)
        #pragma unroll
        for (int nt = 0; nt < 4; ++nt) {
          const int fid = p * 8 + kt * 4 + nt;
          _Float16* dst = &wfrag[fid * 512 + lane * 16];
          const int n = nt * 16 + nl;
          #pragma unroll
          for (int e = 0; e < 16; ++e)
            dst[e] = (_Float16)W1[(p * 64 + kt * 32 + 8 * h + e + (e & 8)) * 64 + n];
        }
    #pragma unroll
    for (int kt = 0; kt < 2; ++kt)
      #pragma unroll
      for (int nt = 0; nt < 2; ++nt) {
        const int fid = 24 + kt * 2 + nt;
        _Float16* dst = &wfrag[fid * 512 + lane * 16];
        const int n = nt * 16 + nl;
        #pragma unroll
        for (int e = 0; e < 16; ++e)
          dst[e] = (_Float16)W2[(kt * 32 + 8 * h + e + (e & 8)) * 32 + n];
      }
    asm volatile("s_wait_dscnt 0" ::: "memory");
  }

  // ---------- stage 1 ----------
  const long b0 = (long)blockIdx.x * 16;

  // A-frag from row-major f32 row (row m = lane&15):
  // elems[0..7] = K = kt*32+8h+0..7 ; elems[8..15] = K = kt*32+16+8h+0..7
  auto load_afrag = [&](const float* row, int kt) -> v16h {
    const float4* f0 = (const float4*)(row + kt * 32 + 8 * h);
    const float4* f1 = (const float4*)(row + kt * 32 + 16 + 8 * h);
    float4 x0 = f0[0], x1 = f0[1], x2 = f1[0], x3 = f1[1];
    v16h r;
    r[0]  = (_Float16)x0.x; r[1]  = (_Float16)x0.y; r[2]  = (_Float16)x0.z; r[3]  = (_Float16)x0.w;
    r[4]  = (_Float16)x1.x; r[5]  = (_Float16)x1.y; r[6]  = (_Float16)x1.z; r[7]  = (_Float16)x1.w;
    r[8]  = (_Float16)x2.x; r[9]  = (_Float16)x2.y; r[10] = (_Float16)x2.z; r[11] = (_Float16)x2.w;
    r[12] = (_Float16)x3.x; r[13] = (_Float16)x3.y; r[14] = (_Float16)x3.z; r[15] = (_Float16)x3.w;
    return r;
  };

  const float* crow = core + (b0 + nl) * 64;
  v16h ca[2];
  ca[0] = load_afrag(crow, 0);
  ca[1] = load_afrag(crow, 1);

  #pragma unroll
  for (int nt = 0; nt < 4; ++nt) {
    v8f acc = {};
    acc = wmma16(ca[0], LDWf(0 + nt), acc);      // W1a kt=0
    acc = wmma16(ca[1], LDWf(4 + nt), acc);      // W1a kt=1
    const float bias = b1[nt * 16 + nl];
    #pragma unroll
    for (int r = 0; r < 8; ++r)                  // C layout: row m = r+8h, col n = nl
      upq[(r + 8 * h) * 64 + nt * 16 + nl] = (_Float16)(acc[r] + bias);
  }

  #pragma unroll 1
  for (int n6 = 0; n6 < 6; ++n6) {
    const float* lrow = ligs + ((b0 + nl) * 6 + n6) * 64;
    v16h la[2];
    la[0] = load_afrag(lrow, 0);
    la[1] = load_afrag(lrow, 1);
    _Float16* Pb = &upq[(1 + n6) * 1024];
    _Float16* Qb = &upq[(7 + n6) * 1024];
    #pragma unroll
    for (int nt = 0; nt < 4; ++nt) {
      v8f accP = {};
      accP = wmma16(la[0], LDWf(8 + nt),  accP); // W1b kt=0
      accP = wmma16(la[1], LDWf(12 + nt), accP); // W1b kt=1
      #pragma unroll
      for (int r = 0; r < 8; ++r)
        Pb[(r + 8 * h) * 64 + nt * 16 + nl] = (_Float16)accP[r];
      v8f accQ = {};
      accQ = wmma16(la[0], LDWf(16 + nt), accQ); // W1c kt=0
      accQ = wmma16(la[1], LDWf(20 + nt), accQ); // W1c kt=1
      #pragma unroll
      for (int r = 0; r < 8; ++r)
        Qb[(r + 8 * h) * 64 + nt * 16 + nl] = (_Float16)accQ[r];
    }
  }
  asm volatile("s_wait_dscnt 0" ::: "memory");

  // ---------- stage 2 ----------
  v16h w2f[2][2];
  #pragma unroll
  for (int kt = 0; kt < 2; ++kt)
    #pragma unroll
    for (int nt = 0; nt < 2; ++nt)
      w2f[kt][nt] = LDWf(24 + kt * 2 + nt);

  const float b2a = b2[nl], b2b = b2[16 + nl];
  const float w3a = W3[nl], w3b = W3[16 + nl];

  float outacc[8];
  #pragma unroll
  for (int r = 0; r < 8; ++r) outacc[r] = 0.0f;

  #pragma unroll 1
  for (int c = 0; c < 30; ++c) {
    const int pbase = (1 + CIa[c]) * 1024;
    const int qbase = (7 + CJa[c]) * 1024;
    v16h af[2];
    #pragma unroll
    for (int kt = 0; kt < 2; ++kt) {
      const int base = nl * 64 + kt * 32 + 8 * h;   // A layout: row m = nl
      v8h u0 = *(const v8h*)&upq[base];
      v8h u1 = *(const v8h*)&upq[base + 16];
      v8h p0 = *(const v8h*)&upq[pbase + base];
      v8h p1 = *(const v8h*)&upq[pbase + base + 16];
      v8h q0 = *(const v8h*)&upq[qbase + base];
      v8h q1 = *(const v8h*)&upq[qbase + base + 16];
      #pragma unroll
      for (int e = 0; e < 8; ++e) {
        af[kt][e]     = (_Float16)softplusf((float)u0[e] + (float)p0[e] + (float)q0[e]);
        af[kt][8 + e] = (_Float16)softplusf((float)u1[e] + (float)p1[e] + (float)q1[e]);
      }
    }
    v8f acc0 = {}, acc1 = {};
    acc0 = wmma16(af[0], w2f[0][0], acc0);
    acc0 = wmma16(af[1], w2f[1][0], acc0);
    acc1 = wmma16(af[0], w2f[0][1], acc1);
    acc1 = wmma16(af[1], w2f[1][1], acc1);
    // layer 3 (N=1): per-row dot with W3 on VALU in C layout
    #pragma unroll
    for (int r = 0; r < 8; ++r)
      outacc[r] += softplusf(acc0[r] + b2a) * w3a + softplusf(acc1[r] + b2b) * w3b;
  }

  // out[b] = 0.5 * sum_combos(dot) + 15 * b3
  const float b3v = b3[0];
  #pragma unroll
  for (int r = 0; r < 8; ++r) {
    float v = outacc[r];
    v += __shfl_xor(v, 8);
    v += __shfl_xor(v, 4);
    v += __shfl_xor(v, 2);
    v += __shfl_xor(v, 1);
    if (nl == 0) out[b0 + r + 8 * h] = 0.5f * v + 15.0f * b3v;
  }
}

extern "C" void kernel_launch(void* const* d_in, const int* in_sizes, int n_in,
                              void* d_out, int out_size, void* d_ws, size_t ws_size,
                              hipStream_t stream) {
  (void)n_in; (void)out_size;
  const float* core = (const float*)d_in[0];
  const float* ligs = (const float*)d_in[1];
  const float* W1   = (const float*)d_in[2];
  const float* b1   = (const float*)d_in[3];
  const float* W2   = (const float*)d_in[4];
  const float* b2   = (const float*)d_in[5];
  const float* W3   = (const float*)d_in[6];
  const float* b3   = (const float*)d_in[7];
  float* out = (float*)d_out;

  const int B = in_sizes[0] / 64;           // 32768
  dim3 grid(B / 16), block(32);             // one wave32 per 16-batch tile

  const size_t WSZ = 28 * 512 * sizeof(_Float16);   // 28 KB of fragments
  if (ws_size >= WSZ) {
    _Float16* wg = (_Float16*)d_ws;
    hipLaunchKernelGGL(swizzle_weights, dim3(28), dim3(32), 0, stream, W1, W2, wg);
    hipLaunchKernelGGL((three_body_kernel<true>), grid, block, 0, stream,
                       core, ligs, W1, b1, W2, b2, W3, b3, wg, out);
  } else {
    hipLaunchKernelGGL((three_body_kernel<false>), grid, block, 0, stream,
                       core, ligs, W1, b1, W2, b2, W3, b3, (const _Float16*)nullptr, out);
  }
}
